// NetworkLocal_60696477827252
// MI455X (gfx1250) — matrix operand order repaired and checked by reading, not compiled, not run on hardware
//
#include <hip/hip_runtime.h>

// ---------------------------------------------------------------------------
// Fused NetworkLocal kernel for MI455X (gfx1250, wave32, WMMA).
//
// Layer-1 GEMM [1M rows x K=29->32 x N=196->208] via v_wmma_f32_16x16x32_f16,
// one WMMA per 16x16 tile (K=32 exactly). Bias folded into the GEMM through
// padded feature col 29 (=1.0) and W0 row 29 (=b0). Mean-pool + leaky-relu
// fused into the WMMA epilogue using lrelu(x) = 0.505 x + 0.495 |x|:
//   - plain sum into a packed float2 accumulator: 4x v_pk_add_f32 on the
//     already pair-aligned WMMA result regs (no movs),
//   - abs sum as a serially dependent scalar chain so |x| folds into VOP3
//     source modifiers (no v_and).
// Layers 2/3 are tiny per-block matrix-vector ops in VALU.
// ---------------------------------------------------------------------------

typedef _Float16 v16h __attribute__((ext_vector_type(16)));
typedef _Float16 v8h  __attribute__((ext_vector_type(8)));
typedef float    v8f  __attribute__((ext_vector_type(8)));
typedef float    v2f  __attribute__((ext_vector_type(2)));

#define NTc   1024
#define NDc   512
#define NBc   6
#define K_IN  29
#define NH    196
#define K_LAT 128
#define KP    32      // padded K (exactly one WMMA K-step); col 29 = 1.0 (bias)
#define NP    208     // padded N (13 tiles of 16)
#define SSTR  40      // s row stride in halfs (80B, 16B-aligned)
#define NTILES 13
#define JTILES 32     // 512 / 16

__device__ __forceinline__ float lrelu(float x) { return fmaxf(x, 0.01f * x); }

__global__ __launch_bounds__(256) void fused_network_local(
    const float* __restrict__ x0, const float* __restrict__ x,
    const float* __restrict__ Nv, const float* __restrict__ basis,
    const float* __restrict__ v,
    const float* __restrict__ W0, const float* __restrict__ b0,
    const float* __restrict__ W1, const float* __restrict__ b1,
    const float* __restrict__ W2, const float* __restrict__ b2,
    float* __restrict__ out)
{
  __shared__ __align__(16) _Float16 sS[NDc * SSTR];   // 40960 B: s rows, f16
  __shared__ __align__(16) _Float16 sW0[KP * NP];     // 13312 B: padded W0+bias, f16
  __shared__ float sHP[NP];                            // pooled hidden
  __shared__ float sH2[NH];                            // layer-2 activations
  __shared__ float sRed[3 * 256];                      // vbulk reduction
  __shared__ float sBas[NBc * 3];
  __shared__ float sVb[3];
  __shared__ _Float16 sCF[16];                         // constant features (f16)

  const int tid  = threadIdx.x;
  const int lane = tid & 31;
  const int wave = tid >> 5;
  const int bi   = blockIdx.x;       // b*NT + i
  const int bb   = bi / NTc;

  // ---- W0 (zero-padded, bias in row 29) into LDS as f16 ----
  for (int idx = tid; idx < KP * NP; idx += 256) sW0[idx] = (_Float16)0.0f;
  if (tid < NBc * 3) sBas[tid] = basis[bb * NBc * 3 + tid];
  __syncthreads();
  for (int idx = tid; idx < K_IN * NH; idx += 256) {
    int k = idx / NH, n = idx - k * NH;
    sW0[k * NP + n] = (_Float16)W0[idx];
  }
  for (int idx = tid; idx < NH; idx += 256)
    sW0[K_IN * NP + idx] = (_Float16)b0[idx];          // row 29 = bias

  // ---- vbulk = mean_j v[b,i,j,:] ----
  float sx = 0.f, sy = 0.f, sz = 0.f;
  for (int j = tid; j < NDc; j += 256) {
    const float* vp = v + ((size_t)bi * NDc + j) * 3;
    sx += vp[0]; sy += vp[1]; sz += vp[2];
  }
  sRed[tid] = sx; sRed[256 + tid] = sy; sRed[512 + tid] = sz;
  __syncthreads();
  for (int off = 128; off > 0; off >>= 1) {
    if (tid < off) {
      sRed[tid]       += sRed[tid + off];
      sRed[256 + tid] += sRed[256 + tid + off];
      sRed[512 + tid] += sRed[512 + tid + off];
    }
    __syncthreads();
  }
  if (tid == 0) {
    float vbx = sRed[0] * (1.0f / NDc);
    float vby = sRed[256] * (1.0f / NDc);
    float vbz = sRed[512] * (1.0f / NDc);
    sVb[0] = vbx; sVb[1] = vby; sVb[2] = vbz;
    const float* p0 = x0 + (size_t)bi * 3;
    float ax = p0[0], ay = p0[1], az = p0[2];
    float x0n = sqrtf(ax * ax + ay * ay + az * az);
    float vbn = sqrtf(vbx * vbx + vby * vby + vbz * vbz);
    sCF[0] = (_Float16)Nv[bi];
    sCF[1] = (_Float16)x0n;
    sCF[8] = (_Float16)vbn;
    for (int k = 0; k < NBc; k++) {
      float bx = sBas[k * 3], by = sBas[k * 3 + 1], bz = sBas[k * 3 + 2];
      sCF[2 + k] = (_Float16)((ax * bx + ay * by + az * bz) / x0n);
      sCF[9 + k] = (_Float16)((vbx * bx + vby * by + vbz * bz) / vbn);
    }
    sCF[15] = (_Float16)0.0f;
  }
  __syncthreads();

  // ---- build s rows (f16): [0..14]=const, 15..28 dynamic, 29=1.0, 30..31=0 ----
  const float vbx = sVb[0], vby = sVb[1], vbz = sVb[2];
  for (int j = tid; j < NDc; j += 256) {
    const float* xp = x + ((size_t)bi * NDc + j) * 3;
    const float* vp = v + ((size_t)bi * NDc + j) * 3;
    float xa = xp[0], xb = xp[1], xc = xp[2];
    float va = vp[0] - vbx, vb = vp[1] - vby, vc = vp[2] - vbz;
    float xn = sqrtf(xa * xa + xb * xb + xc * xc);
    float vn = sqrtf(va * va + vb * vb + vc * vc);
    _Float16* row = &sS[j * SSTR];
    #pragma unroll
    for (int t = 0; t < 15; t++) row[t] = sCF[t];
    row[15] = (_Float16)xn;
    row[22] = (_Float16)vn;
    #pragma unroll
    for (int k = 0; k < NBc; k++) {
      float bx = sBas[k * 3], by = sBas[k * 3 + 1], bz = sBas[k * 3 + 2];
      row[16 + k] = (_Float16)((xa * bx + xb * by + xc * bz) / xn);
      row[23 + k] = (_Float16)((va * bx + vb * by + vc * bz) / vn);
    }
    row[29] = (_Float16)1.0f;                          // bias feature
    row[30] = (_Float16)0.0f;
    row[31] = (_Float16)0.0f;
  }
  __syncthreads();

  // ---- layer-1 GEMM + mean pool via WMMA f32_16x16x32_f16 ----
  // A lane L: M = L%16, halfs e<8 -> K = e + (L<16?0:8), e>=8 -> K = e+8 + (L<16?0:8)
  // B lane L: N = L%16, halfs e<8 -> K = e + (L<16?0:8), e>=8 -> K = e+8 + (L<16?0:8)
  // C lane L: c[m] = H[M = m + (L<16?0:8)][N = L%16]
  const int kbaseA = (lane < 16) ? 0 : 8;
  const int kbaseB = (lane < 16) ? 0 : 8;
  const int lcol   = lane & 15;

  for (int nt = wave; nt < NTILES; nt += 8) {
    const int col = nt * 16 + lcol;
    v16h bf;
    #pragma unroll
    for (int e = 0; e < 16; e++) bf[e] = sW0[(kbaseB + e + (e & 8)) * NP + col];
    v2f  accS2 = {0.0f, 0.0f};   // packed running sum of h (pk_add_f32)
    float accA = 0.0f;           // running sum of |h|
    for (int jt = 0; jt < JTILES; jt++) {
      const _Float16* rowp = &sS[(jt * 16 + lcol) * SSTR];
      v8h lo = *(const v8h*)(rowp + kbaseA);        // K = kA .. kA+7
      v8h hi = *(const v8h*)(rowp + kbaseA + 16);   // K = kA+16 .. kA+23
      v16h af = __builtin_shufflevector(lo, hi, 0, 1, 2, 3, 4, 5, 6, 7,
                                                8, 9, 10, 11, 12, 13, 14, 15);
      v8f c = {};
      c = __builtin_amdgcn_wmma_f32_16x16x32_f16(
          /*neg_a=*/false, af, /*neg_b=*/false, bf,
          /*c_mod=*/(short)0, c, /*reuse_a=*/false, /*reuse_b=*/false);
      // Packed sum: 4x v_pk_add_f32 on pair-aligned WMMA result registers.
      accS2 += __builtin_shufflevector(c, c, 0, 1);
      accS2 += __builtin_shufflevector(c, c, 2, 3);
      accS2 += __builtin_shufflevector(c, c, 4, 5);
      accS2 += __builtin_shufflevector(c, c, 6, 7);
      // Abs sum: serially dependent chain -> abs folds into VOP3 src mods.
      float as = accA;
      as += __builtin_fabsf(c[0]);
      as += __builtin_fabsf(c[1]);
      as += __builtin_fabsf(c[2]);
      as += __builtin_fabsf(c[3]);
      as += __builtin_fabsf(c[4]);
      as += __builtin_fabsf(c[5]);
      as += __builtin_fabsf(c[6]);
      as += __builtin_fabsf(c[7]);
      accA = as;
    }
    // lrelu(x) = 0.505 x + 0.495 |x|  (slope 0.01); apply once per column.
    float accS = accS2[0] + accS2[1];
    float pooled = 0.505f * accS + 0.495f * accA;
    pooled += __shfl_xor(pooled, 16, 32);           // combine M 0..7 with 8..15
    if (lane < 16) sHP[col] = pooled * (1.0f / NDc);
  }
  __syncthreads();

  // ---- layer 2: h2 = lrelu(hpool @ W1 + b1) ----
  if (tid < NH) {
    float acc = b1[tid];
    for (int k = 0; k < NH; k++) acc += sHP[k] * W1[k * NH + tid];
    sH2[tid] = lrelu(acc);
  }
  __syncthreads();

  // ---- layer 3: out = h2 @ W2 + b2 ----
  if (tid < K_LAT) {
    float acc = b2[tid];
    for (int k = 0; k < NH; k++) acc += sH2[k] * W2[k * K_LAT + tid];
    out[(size_t)bi * K_LAT + tid] = acc;
  }
}

extern "C" void kernel_launch(void* const* d_in, const int* in_sizes, int n_in,
                              void* d_out, int out_size, void* d_ws, size_t ws_size,
                              hipStream_t stream) {
  const float* x0    = (const float*)d_in[0];
  const float* x     = (const float*)d_in[1];
  const float* Nv    = (const float*)d_in[2];
  const float* basis = (const float*)d_in[3];
  const float* v     = (const float*)d_in[4];
  const float* W0    = (const float*)d_in[5];
  const float* b0    = (const float*)d_in[6];
  const float* W1    = (const float*)d_in[7];
  const float* b1    = (const float*)d_in[8];
  const float* W2    = (const float*)d_in[9];
  const float* b2    = (const float*)d_in[10];
  float* out = (float*)d_out;

  const int BNT = in_sizes[2];  // B * NT = 2048 (one block per (b,i))
  fused_network_local<<<BNT, 256, 0, stream>>>(
      x0, x, Nv, basis, v, W0, b0, W1, b1, W2, b2, out);
}
